// BlankAttention_57913339019596
// MI455X (gfx1250) — matrix operand order repaired and checked by reading, not compiled, not run on hardware
//
#include <hip/hip_runtime.h>
#include <cstdint>
#include <cstddef>

// ---------------------------------------------------------------------------
// MI455X (gfx1250) causal attention block, bf16 WMMA everywhere.
//   kernel 1: QKV projection GEMM  (fp32 in -> bf16 q/k/v workspace)
//   kernel 2: flash attention      (bf16 in -> bf16 O workspace)
//   kernel 3: output projection    (bf16 O  -> fp32 d_out)
// GEMMs: workgroup tile 128x128, wave tile 16x128, K-step 32, 2-stage
// global->LDS software pipeline.
// ---------------------------------------------------------------------------

typedef __bf16 bf16;
typedef __attribute__((ext_vector_type(16))) bf16  v16bf;
typedef __attribute__((ext_vector_type(8)))  bf16  bf16x8;
typedef __attribute__((ext_vector_type(8)))  float v8f;

constexpr int Sq    = 2048;           // sequence length
constexpr int DM    = 2048;           // model dim
constexpr int H     = 16;             // heads
constexpr int DH    = 128;            // head dim
constexpr int NQKV  = 3 * H * DH;     // 6144
constexpr int Bn    = 2;              // batch

__device__ __forceinline__ bf16 f2bf(float f) { return (bf16)f; }

__device__ __forceinline__ v16bf make_v16(bf16x8 lo, bf16x8 hi) {
  v16bf r;
#pragma unroll
  for (int i = 0; i < 8; ++i) { r[i] = lo[i]; r[i + 8] = hi[i]; }
  return r;
}

__device__ __forceinline__ v8f wmma_bf16(v16bf a, v16bf b, v8f c) {
  return __builtin_amdgcn_wmma_f32_16x16x32_bf16(
      /*neg_a=*/false, a, /*neg_b=*/false, b,
      /*c_mod=*/(short)0, c, /*reuse_a=*/false, /*reuse_b=*/false);
}

// A fragment (16x32 bf16) from row-major LDS tile with element stride `stride`.
// lane L (L<16): row=L, K chunks [0..7]+[16..23]; lane L+16: row=L, [8..15]+[24..31].
__device__ __forceinline__ v16bf load_afrag(const bf16* base, int stride, int lr, int hi) {
  const bf16* p = base + lr * stride + hi * 8;
  return make_v16(*(const bf16x8*)p, *(const bf16x8*)(p + 16));
}

// B fragment (32x16, K x N) from N-major (transposed) LDS tile: row = N, K contiguous.
// lane L (L<16): col N=L, K chunks [0..7]+[16..23]; lane L+16: col N=L, [8..15]+[24..31].
__device__ __forceinline__ v16bf load_bfrag(const bf16* base, int stride, int lr, int hi) {
  const bf16* p = base + lr * stride + hi * 8;
  return make_v16(*(const bf16x8*)p, *(const bf16x8*)(p + 16));
}

// ---------------------------------------------------------------------------
// Kernel 1: QKV projection.  C[4096 x 6144] = X[4096 x 2048] @ Win[2048 x 6144]
// ---------------------------------------------------------------------------
__global__ __launch_bounds__(256) void qkv_gemm_kernel(
    const float* __restrict__ X, const float* __restrict__ W,
    bf16* __restrict__ Qb, bf16* __restrict__ Kb, bf16* __restrict__ Vb) {
  __shared__ alignas(16) bf16 As[128 * 40];   // A tile, row-major, padded
  __shared__ alignas(16) bf16 Bt[128 * 40];   // B tile transposed (N-major)

  const int t    = threadIdx.x;
  const int wave = t >> 5, lane = t & 31, lr = lane & 15, hi = lane >> 4;
  const int m0 = blockIdx.y * 128;
  const int n0 = blockIdx.x * 128;

  // staging slots (global -> regs -> LDS pipeline)
  const int arow = t >> 1, ahalf = t & 1;           // A: 128 rows x 2 halves
  const int bk   = t >> 3, bseg  = t & 7;           // B: 32 k-rows x 8 segs of 16
  const float* aSrcBase = X + (size_t)(m0 + arow) * DM + ahalf * 16;
  const float* bSrcBase = W + (size_t)bk * NQKV + n0 + bseg * 16;

  float4 a0, a1, a2, a3, b0, b1, b2, b3;
  { const float* s = aSrcBase;
    a0 = *(const float4*)(s + 0);  a1 = *(const float4*)(s + 4);
    a2 = *(const float4*)(s + 8);  a3 = *(const float4*)(s + 12); }
  { const float* s = bSrcBase;
    b0 = *(const float4*)(s + 0);  b1 = *(const float4*)(s + 4);
    b2 = *(const float4*)(s + 8);  b3 = *(const float4*)(s + 12); }

  v8f acc[8] = {};

  for (int k0 = 0; k0 < DM; k0 += 32) {
    { // store staged A (fp32 -> bf16)
      bf16* dst = As + arow * 40 + ahalf * 16;
      dst[0]=f2bf(a0.x); dst[1]=f2bf(a0.y); dst[2]=f2bf(a0.z); dst[3]=f2bf(a0.w);
      dst[4]=f2bf(a1.x); dst[5]=f2bf(a1.y); dst[6]=f2bf(a1.z); dst[7]=f2bf(a1.w);
      dst[8]=f2bf(a2.x); dst[9]=f2bf(a2.y); dst[10]=f2bf(a2.z); dst[11]=f2bf(a2.w);
      dst[12]=f2bf(a3.x); dst[13]=f2bf(a3.y); dst[14]=f2bf(a3.z); dst[15]=f2bf(a3.w);
    }
    { // store staged B transposed (fp32 -> bf16)
      const int nb = bseg * 16;
      Bt[(nb+0)*40+bk]=f2bf(b0.x);  Bt[(nb+1)*40+bk]=f2bf(b0.y);
      Bt[(nb+2)*40+bk]=f2bf(b0.z);  Bt[(nb+3)*40+bk]=f2bf(b0.w);
      Bt[(nb+4)*40+bk]=f2bf(b1.x);  Bt[(nb+5)*40+bk]=f2bf(b1.y);
      Bt[(nb+6)*40+bk]=f2bf(b1.z);  Bt[(nb+7)*40+bk]=f2bf(b1.w);
      Bt[(nb+8)*40+bk]=f2bf(b2.x);  Bt[(nb+9)*40+bk]=f2bf(b2.y);
      Bt[(nb+10)*40+bk]=f2bf(b2.z); Bt[(nb+11)*40+bk]=f2bf(b2.w);
      Bt[(nb+12)*40+bk]=f2bf(b3.x); Bt[(nb+13)*40+bk]=f2bf(b3.y);
      Bt[(nb+14)*40+bk]=f2bf(b3.z); Bt[(nb+15)*40+bk]=f2bf(b3.w);
    }
    __syncthreads();

    if (k0 + 32 < DM) { // issue next-step global loads; overlap with compute
      const float* s = aSrcBase + (k0 + 32);
      a0 = *(const float4*)(s + 0);  a1 = *(const float4*)(s + 4);
      a2 = *(const float4*)(s + 8);  a3 = *(const float4*)(s + 12);
      const float* s2 = bSrcBase + (size_t)(k0 + 32) * NQKV;
      b0 = *(const float4*)(s2 + 0); b1 = *(const float4*)(s2 + 4);
      b2 = *(const float4*)(s2 + 8); b3 = *(const float4*)(s2 + 12);
    }

    v16bf afrag = load_afrag(As + wave * 16 * 40, 40, lr, hi);
#pragma unroll
    for (int n = 0; n < 8; ++n) {
      v16bf bfrag = load_bfrag(Bt + n * 16 * 40, 40, lr, hi);
      acc[n] = wmma_bf16(afrag, bfrag, acc[n]);
    }
    __syncthreads();
  }

  // Epilogue: fragment-uniform head / qkv-select / d-base (scalar branches)
  const int mbase = m0 + wave * 16;
#pragma unroll
  for (int n = 0; n < 8; ++n) {
    const int cb    = n0 + n * 16;       // uniform across lanes
    const int hh    = cb / 384;
    const int rr    = cb - hh * 384;
    const int which = rr >> 7;
    const int dbase = rr & 127;
    bf16* dst;
    float scl = 1.0f;
    if (which == 0)      { dst = Qb; scl = 0.08838834764831845f; }
    else if (which == 1) { dst = Kb; }
    else                 { dst = Vb; }
#pragma unroll
    for (int r = 0; r < 8; ++r) {
      const int m = mbase + r + hi * 8;
      const int bidx = m >> 11;          // / 2048
      const int s = m & 2047;
      const size_t off = (((size_t)(bidx * H + hh)) * Sq + s) * DH + dbase + lr;
      dst[off] = f2bf(acc[n][r] * scl);
    }
  }
}

// ---------------------------------------------------------------------------
// Kernel 2: flash attention per (b,h).  Workgroup = 128 queries, wave = 16.
// ---------------------------------------------------------------------------
__global__ __launch_bounds__(256) void attn_kernel(
    const bf16* __restrict__ Qb, const bf16* __restrict__ Kb,
    const bf16* __restrict__ Vb, bf16* __restrict__ Ob) {
  __shared__ alignas(16) bf16 Kblk[32 * 136];    // [key][d], padded
  __shared__ alignas(16) bf16 Vt[128 * 40];      // [d][key], padded
  __shared__ alignas(16) bf16 Plds[8 * 16 * 40]; // per-wave 16x32 P tile

  const int t    = threadIdx.x;
  const int wave = t >> 5, lane = t & 31, lr = lane & 15, hi = lane >> 4;
  const int bh  = blockIdx.y;
  const int b   = bh >> 4, h = bh & 15;
  const int q0w = blockIdx.x * 128;

  const bf16* Qh = Qb + (size_t)bh * Sq * DH;
  const bf16* Kh = Kb + (size_t)bh * Sq * DH;
  const bf16* Vh = Vb + (size_t)bh * Sq * DH;

  // Load Q fragments: 16 rows x 128 d -> 4 A fragments of 16x32
  const int qrow = q0w + wave * 16 + lr;
  v16bf qf[4];
#pragma unroll
  for (int f = 0; f < 4; ++f) {
    const bf16* qp = Qh + (size_t)qrow * DH + f * 32 + hi * 8;
    qf[f] = make_v16(*(const bf16x8*)qp, *(const bf16x8*)(qp + 16));
  }

  v8f  o_acc[8] = {};
  float mrow[8], lrow[8];
#pragma unroll
  for (int r = 0; r < 8; ++r) { mrow[r] = -INFINITY; lrow[r] = 0.0f; }

  const int q_tile_end = q0w + wave * 16 + 15;
  const int jend = q0w + 128;
  constexpr float LOG2E = 1.4426950408889634f;

  const int krow = t >> 3, kseg = t & 7;   // cooperative-load mapping

  for (int j0 = 0; j0 < jend; j0 += 32) {
    { // cooperative K block load: 32 keys x 128 d, row-major
      const uint4* src = (const uint4*)(Kh + (size_t)(j0 + krow) * DH + kseg * 16);
      uint4 d0 = src[0], d1 = src[1];
      uint4* dst = (uint4*)(Kblk + krow * 136 + kseg * 16);
      dst[0] = d0; dst[1] = d1;
    }
    { // cooperative V block load, transposed into [d][key]
      const bf16* src = Vh + (size_t)(j0 + krow) * DH + kseg * 16;
      bf16x8 v0 = *(const bf16x8*)src;
      bf16x8 v1 = *((const bf16x8*)src + 1);
#pragma unroll
      for (int i = 0; i < 8; ++i) {
        Vt[(kseg * 16 + i) * 40 + krow]     = v0[i];
        Vt[(kseg * 16 + 8 + i) * 40 + krow] = v1[i];
      }
    }
    if (j0 + 32 < jend) { // prefetch next key block (global_prefetch_b8)
      __builtin_prefetch(Kh + (size_t)(j0 + 32 + krow) * DH + kseg * 16, 0, 1);
      __builtin_prefetch(Vh + (size_t)(j0 + 32 + krow) * DH + kseg * 16, 0, 1);
    }
    __syncthreads();

    if (j0 <= q_tile_end) {   // wave-uniform: EXEC stays all-ones for WMMA
      // --- scores: S = Q @ K^T, two 16-key fragments ---
      v8f sc[2] = {};
#pragma unroll
      for (int f2 = 0; f2 < 2; ++f2) {
#pragma unroll
        for (int dc = 0; dc < 4; ++dc) {
          v16bf kb = load_bfrag(Kblk + f2 * 16 * 136 + dc * 32, 136, lr, hi);
          sc[f2] = wmma_bf16(qf[dc], kb, sc[f2]);
        }
      }

      // --- causal mask + online softmax (rows live in 16-lane halves) ---
#pragma unroll
      for (int r = 0; r < 8; ++r) {
        const int q  = q0w + wave * 16 + r + hi * 8;
        float s0 = (j0 + lr      > q) ? -INFINITY : sc[0][r];
        float s1 = (j0 + 16 + lr > q) ? -INFINITY : sc[1][r];
        float mx = fmaxf(s0, s1);
#pragma unroll
        for (int off = 1; off < 16; off <<= 1)
          mx = fmaxf(mx, __shfl_xor(mx, off, 32));
        const float mold = mrow[r];
        const float mnew = fmaxf(mold, mx);
        const float corr = (mold == -INFINITY) ? 0.0f : exp2f((mold - mnew) * LOG2E);
        const float p0 = exp2f((s0 - mnew) * LOG2E);  // -inf -> 0
        const float p1 = exp2f((s1 - mnew) * LOG2E);
        float ps = p0 + p1;
#pragma unroll
        for (int off = 1; off < 16; off <<= 1)
          ps += __shfl_xor(ps, off, 32);
        lrow[r] = lrow[r] * corr + ps;
        mrow[r] = mnew;
#pragma unroll
        for (int n = 0; n < 8; ++n) o_acc[n][r] *= corr;
        sc[0][r] = p0; sc[1][r] = p1;
      }

      // --- re-layout P (C-frag) -> A-frag via wave-private LDS ---
      bf16* pw = Plds + wave * 16 * 40;
#pragma unroll
      for (int r = 0; r < 8; ++r) {
        const int row = r + hi * 8;
        pw[row * 40 + lr]      = f2bf(sc[0][r]);
        pw[row * 40 + 16 + lr] = f2bf(sc[1][r]);
      }
      v16bf pfrag = load_afrag(pw, 40, lr, hi);   // same-wave LDS RAW

      // --- O += P @ V (contraction over 32 keys, 8 d-column fragments) ---
#pragma unroll
      for (int n = 0; n < 8; ++n) {
        v16bf vb = load_bfrag(Vt + n * 16 * 40, 40, lr, hi);
        o_acc[n] = wmma_bf16(pfrag, vb, o_acc[n]);
      }
    }
    __syncthreads();
  }

  // Epilogue: normalize, store O[b*S+q, h*128+d] as bf16
#pragma unroll
  for (int r = 0; r < 8; ++r) {
    const float inv = 1.0f / lrow[r];
    const int q = q0w + wave * 16 + r + hi * 8;
    const size_t rowoff = ((size_t)(b * Sq + q)) * DM + h * DH;
#pragma unroll
    for (int n = 0; n < 8; ++n)
      Ob[rowoff + n * 16 + lr] = f2bf(o_acc[n][r] * inv);
  }
}

// ---------------------------------------------------------------------------
// Kernel 3: output projection.  Y[4096 x 2048] = O(bf16) @ Wout(fp32->bf16)
// ---------------------------------------------------------------------------
__global__ __launch_bounds__(256) void out_gemm_kernel(
    const bf16* __restrict__ A, const float* __restrict__ W,
    float* __restrict__ out) {
  __shared__ alignas(16) bf16 As[128 * 40];
  __shared__ alignas(16) bf16 Bt[128 * 40];

  const int t    = threadIdx.x;
  const int wave = t >> 5, lane = t & 31, lr = lane & 15, hi = lane >> 4;
  const int m0 = blockIdx.y * 128;
  const int n0 = blockIdx.x * 128;

  const int arow = t >> 1, ahalf = t & 1;
  const int bk   = t >> 3, bseg  = t & 7;
  const bf16*  aSrcBase = A + (size_t)(m0 + arow) * DM + ahalf * 16;
  const float* bSrcBase = W + (size_t)bk * DM + n0 + bseg * 16;

  uint4 a0, a1;
  float4 b0, b1, b2, b3;
  { const uint4* s = (const uint4*)aSrcBase; a0 = s[0]; a1 = s[1]; }
  { const float* s = bSrcBase;
    b0 = *(const float4*)(s + 0); b1 = *(const float4*)(s + 4);
    b2 = *(const float4*)(s + 8); b3 = *(const float4*)(s + 12); }

  v8f acc[8] = {};

  for (int k0 = 0; k0 < DM; k0 += 32) {
    { uint4* dst = (uint4*)(As + arow * 40 + ahalf * 16);
      dst[0] = a0; dst[1] = a1; }
    { const int nb = bseg * 16;
      Bt[(nb+0)*40+bk]=f2bf(b0.x);  Bt[(nb+1)*40+bk]=f2bf(b0.y);
      Bt[(nb+2)*40+bk]=f2bf(b0.z);  Bt[(nb+3)*40+bk]=f2bf(b0.w);
      Bt[(nb+4)*40+bk]=f2bf(b1.x);  Bt[(nb+5)*40+bk]=f2bf(b1.y);
      Bt[(nb+6)*40+bk]=f2bf(b1.z);  Bt[(nb+7)*40+bk]=f2bf(b1.w);
      Bt[(nb+8)*40+bk]=f2bf(b2.x);  Bt[(nb+9)*40+bk]=f2bf(b2.y);
      Bt[(nb+10)*40+bk]=f2bf(b2.z); Bt[(nb+11)*40+bk]=f2bf(b2.w);
      Bt[(nb+12)*40+bk]=f2bf(b3.x); Bt[(nb+13)*40+bk]=f2bf(b3.y);
      Bt[(nb+14)*40+bk]=f2bf(b3.z); Bt[(nb+15)*40+bk]=f2bf(b3.w); }
    __syncthreads();

    if (k0 + 32 < DM) {
      const uint4* s = (const uint4*)(aSrcBase + (k0 + 32));
      a0 = s[0]; a1 = s[1];
      const float* s2 = bSrcBase + (size_t)(k0 + 32) * DM;
      b0 = *(const float4*)(s2 + 0); b1 = *(const float4*)(s2 + 4);
      b2 = *(const float4*)(s2 + 8); b3 = *(const float4*)(s2 + 12);
    }

    v16bf afrag = load_afrag(As + wave * 16 * 40, 40, lr, hi);
#pragma unroll
    for (int n = 0; n < 8; ++n) {
      v16bf bfrag = load_bfrag(Bt + n * 16 * 40, 40, lr, hi);
      acc[n] = wmma_bf16(afrag, bfrag, acc[n]);
    }
    __syncthreads();
  }

  const int mbase = m0 + wave * 16;
#pragma unroll
  for (int n = 0; n < 8; ++n) {
#pragma unroll
    for (int r = 0; r < 8; ++r) {
      const int m = mbase + r + hi * 8;
      const int c = n0 + n * 16 + lr;
      out[(size_t)m * DM + c] = acc[n][r];
    }
  }
}

// ---------------------------------------------------------------------------
extern "C" void kernel_launch(void* const* d_in, const int* in_sizes, int n_in,
                              void* d_out, int out_size, void* d_ws, size_t ws_size,
                              hipStream_t stream) {
  (void)in_sizes; (void)n_in; (void)out_size; (void)ws_size;
  const float* x     = (const float*)d_in[0];
  const float* w_in  = (const float*)d_in[1];
  const float* w_out = (const float*)d_in[2];
  // d_in[3] (mask) unused: causal mask computed analytically.

  const size_t HS = (size_t)Bn * H * Sq * DH;   // 8,388,608 elements
  bf16* Qb = (bf16*)d_ws;
  bf16* Kb = Qb + HS;
  bf16* Vb = Kb + HS;
  bf16* Ob = Vb + HS;                            // [B*S, H*DH] bf16
  float* y = (float*)d_out;

  qkv_gemm_kernel<<<dim3(NQKV / 128, (Bn * Sq) / 128), 256, 0, stream>>>(
      x, w_in, Qb, Kb, Vb);
  attn_kernel<<<dim3(Sq / 128, Bn * H), 256, 0, stream>>>(Qb, Kb, Vb, Ob);
  out_gemm_kernel<<<dim3(DM / 128, (Bn * Sq) / 128), 256, 0, stream>>>(
      Ob, w_out, y);
}
